// Tucker_7507602833579
// MI455X (gfx1250) — matrix operand order repaired and checked by reading, not compiled, not run on hardware
//
#include <hip/hip_runtime.h>
#include <hip/hip_bf16.h>
#include <math.h>

// ---------------- types & helpers ----------------
typedef __attribute__((ext_vector_type(16))) __bf16 v16bf;
typedef __attribute__((ext_vector_type(8)))  float  v8f;

union FragA { v16bf v; uint4 q[2]; unsigned int w[8]; unsigned short u[16]; };
union FragC { v8f   v; float f[8]; };

__device__ __forceinline__ unsigned short f2bf(float f) {
    unsigned int u = __float_as_uint(f);
    u += 0x7FFFu + ((u >> 16) & 1u);     // round-to-nearest-even
    return (unsigned short)(u >> 16);
}
__device__ __forceinline__ float bf2f(unsigned short h) {
    return __uint_as_float(((unsigned int)h) << 16);
}
// packed bf16 multiply: 2 elements per instruction (V_PK_MUL_BF16, VOP3P op 42)
__device__ __forceinline__ unsigned int pkmul_bf16(unsigned int a, unsigned int b) {
    unsigned int d;
    asm volatile("v_pk_mul_bf16 %0, %1, %2" : "=v"(d) : "v"(a), "v"(b));
    return d;
}
// packed f32 -> bf16 convert (RNE): dst = {hi=bf16(h), lo=bf16(l)}
__device__ __forceinline__ unsigned int cvt_pk_bf16(float l, float h) {
    unsigned int d;
    asm volatile("v_cvt_pk_bf16_f32 %0, %1, %2" : "=v"(d) : "v"(l), "v"(h));
    return d;
}

#define Bsz 1024
#define Dd  256
#define EPS 1e-5f

// ---------------- BN0: stats over gathered batch ----------------
__global__ __launch_bounds__(256) void prep0_stats(
    const int* __restrict__ e1, const float* __restrict__ E,
    const float* __restrict__ g0, const float* __restrict__ b0,
    float* __restrict__ scale0, float* __restrict__ shift0)
{
    const int d = threadIdx.x;
    float s = 0.f, s2 = 0.f;
    for (int b = 0; b < Bsz; ++b) {
        float v = E[(size_t)e1[b] * Dd + d];
        s += v; s2 += v * v;
    }
    float mean = s * (1.f / Bsz);
    float var  = s2 * (1.f / Bsz) - mean * mean;
    float istd = rsqrtf(var + EPS);
    float sc   = istd * g0[d];
    scale0[d] = sc;
    shift0[d] = b0[d] - mean * sc;
}

// ---------------- BN0 apply + gathers (x0 bf16, r f32) ----------------
__global__ __launch_bounds__(256) void prep0_apply(
    const int* __restrict__ e1, const int* __restrict__ rel,
    const float* __restrict__ E, const float* __restrict__ R,
    const float* __restrict__ scale0, const float* __restrict__ shift0,
    unsigned short* __restrict__ x0, float* __restrict__ rbuf)
{
    const int b = blockIdx.x, d = threadIdx.x;
    float v = E[(size_t)e1[b] * Dd + d];
    x0[(size_t)b * Dd + d]   = f2bf(v * scale0[d] + shift0[d]);
    rbuf[(size_t)b * Dd + d] = R[(size_t)rel[b] * Dd + d];
}

// ---------------- GEMM1: [1024 x 65536] x [65536 x 256], split-K=8 ----------------
// A[b, d*256+i] = r[b,d]*x0[b,i] built in registers (pk_mul_bf16 broadcast scale);
// B = W flattened [65536,256], staged double-buffered bf16 in LDS via v_cvt_pk_bf16_f32.
__global__ __launch_bounds__(256) void gemm1_kernel(
    const float* __restrict__ Wf, const unsigned short* __restrict__ x0,
    const float* __restrict__ rbuf, float* __restrict__ out1p)
{
    extern __shared__ char smem[];
    unsigned short* ldsX = (unsigned short*)smem;       // 128*256 halfs (64KB)
    unsigned short* ldsB = ldsX + 128 * Dd;             // 2 x (64 n-rows x 32 k) (8KB)

    const int tid  = threadIdx.x;
    const int lane = tid & 31, wave = tid >> 5;
    const int nbase = blockIdx.x * 64;
    const int mbase = blockIdx.y * 128;
    const int zc    = blockIdx.z;                       // K chunk 0..7

    // Stage x0 M-tile once (fully coalesced b128 copies)
    {
        const uint4* src = (const uint4*)(x0 + (size_t)mbase * Dd);
        uint4* dst = (uint4*)ldsX;
        for (int i = tid; i < 128 * Dd / 8; i += 256) dst[i] = src[i];
    }
    __syncthreads();

    FragC acc[4];
#pragma unroll
    for (int s = 0; s < 4; ++s)
#pragma unroll
        for (int i = 0; i < 8; ++i) acc[s].f[i] = 0.f;

    const int ml = (wave << 4) + (lane & 15);
    const unsigned short* xrow = ldsX + ml * Dd;
    const int ahi = (lane & 16) ? 8 : 0;
    const int bhi = (lane & 16) ? 8 : 0;
    const int tn = tid & 63, tg = tid >> 6;             // tg 0..3: owns kl pairs

    const int kkStart = zc * 256, kkEnd = kkStart + 256;

    // each thread: 4 adjacent-k pairs {2*tg+8j, 2*tg+8j+1}, j=0..3
    float wreg[8];
    auto do_loads = [&](int kk) {
        const float* wp = Wf + (size_t)(kk << 5) * Dd + nbase + tn;
        __builtin_prefetch((const void*)(wp + 32 * Dd), 0, 1); // next k-block
#pragma unroll
        for (int j = 0; j < 4; ++j) {
            int kl = 2 * tg + (j << 3);
            wreg[2 * j]     = wp[(size_t)kl * Dd];
            wreg[2 * j + 1] = wp[(size_t)(kl + 1) * Dd];
        }
    };

    do_loads(kkStart);                 // prologue
    unsigned int rpack = 0;

    for (int kk = kkStart; kk < kkEnd; ++kk) {
        const int i0 = (kk & 7) << 5;  // i offset within x0 row
        unsigned short* bb = ldsB + (kk & 1) * (64 * 32);

        // packed convert + b32 store current W regs -> LDS ([n][k] bf16)
#pragma unroll
        for (int j = 0; j < 4; ++j) {
            int kl = 2 * tg + (j << 3);
            *(unsigned int*)(bb + tn * 32 + kl) =
                cvt_pk_bf16(wreg[2 * j], wreg[2 * j + 1]);
        }

        // issue next step's global W loads (overlaps with this step's WMMAs)
        if (kk + 1 < kkEnd) do_loads(kk + 1);

        if ((kk & 7) == 0) {
            unsigned short rb = f2bf(rbuf[(size_t)(mbase + ml) * Dd + (kk >> 3)]);
            rpack = ((unsigned int)rb << 16) | rb;
        }
        __syncthreads();

        // Build A fragment: bf16(r) * bf16(x0), ISA 16-bit A 16x32 layout
        FragA a;
        {
            const uint4* p0 = (const uint4*)(xrow + i0 + ahi);
            a.q[0] = p0[0];            // K = i0+ahi .. +7
            a.q[1] = p0[2];            // K = i0+ahi+16 .. +23
#pragma unroll
            for (int t = 0; t < 8; ++t) a.w[t] = pkmul_bf16(a.w[t], rpack);
        }

        // Batch all B-fragment DS loads, then issue 4 WMMAs
        FragA bf4[4];
#pragma unroll
        for (int s = 0; s < 4; ++s) {
            const uint4* bp = (const uint4*)(bb + (s * 16 + (lane & 15)) * 32 + bhi);
            bf4[s].q[0] = bp[0]; bf4[s].q[1] = bp[2];
        }
#pragma unroll
        for (int s = 0; s < 4; ++s)
            acc[s].v = __builtin_amdgcn_wmma_f32_16x16x32_bf16(
                false, a.v, false, bf4[s].v, (short)0, acc[s].v, false, false);
    }

    // Store split-K partial (C layout: lanes 0-15 N=lane M=v; 16-31 N=lane-16 M=v+8)
    float* outp = out1p + (size_t)zc * Bsz * Dd;
#pragma unroll
    for (int s = 0; s < 4; ++s) {
        int n = nbase + s * 16 + (lane & 15);
#pragma unroll
        for (int v = 0; v < 8; ++v) {
            int m = mbase + (wave << 4) + v + ((lane & 16) ? 8 : 0);
            outp[(size_t)m * Dd + n] = acc[s].f[v];
        }
    }
}

// ---------------- BN1: reduce split-K partials + stats ----------------
__global__ __launch_bounds__(256) void bn1_stats(
    const float* __restrict__ out1p, const float* __restrict__ g1,
    const float* __restrict__ b1, float* __restrict__ out1,
    float* __restrict__ scale1, float* __restrict__ shift1)
{
    const int d = threadIdx.x;
    float s = 0.f, s2 = 0.f;
    for (int b = 0; b < Bsz; ++b) {
        float v = 0.f;
#pragma unroll
        for (int z = 0; z < 8; ++z)
            v += out1p[((size_t)z * Bsz + b) * Dd + d];
        out1[(size_t)b * Dd + d] = v;
        s += v; s2 += v * v;
    }
    float mean = s * (1.f / Bsz);
    float var  = s2 * (1.f / Bsz) - mean * mean;
    float istd = rsqrtf(var + EPS);
    float sc   = istd * g1[d];
    scale1[d] = sc;
    shift1[d] = b1[d] - mean * sc;
}

__global__ __launch_bounds__(256) void bn1_apply(
    const float* __restrict__ out1, const float* __restrict__ scale1,
    const float* __restrict__ shift1, unsigned short* __restrict__ x1)
{
    const int b = blockIdx.x, d = threadIdx.x;
    x1[(size_t)b * Dd + d] = f2bf(out1[(size_t)b * Dd + d] * scale1[d] + shift1[d]);
}

// ---------------- GEMM2: logits = x1 @ E^T, fused sigmoid ----------------
// E rows are already the [n][k] layout WMMA-B wants; packed-convert f32->bf16 into LDS.
__global__ __launch_bounds__(256) void gemm2_kernel(
    const float* __restrict__ E, const unsigned short* __restrict__ x1,
    float* __restrict__ out, int NE)
{
    extern __shared__ char smem[];
    unsigned short* ldsB = (unsigned short*)smem;       // 128 rows x 256 halfs (64KB)

    const int tid  = threadIdx.x;
    const int lane = tid & 31, wave = tid >> 5;
    const int nbase = blockIdx.x * 128;
    const int mbase = blockIdx.y * 128;

    // Stage 128 entity rows (bounds-checked): b64 loads + packed convert + b32 stores
    for (int i = tid; i < 128 * Dd / 2; i += 256) {
        int n = i >> 7, k2 = (i & 127) * 2;
        int ng = nbase + n;
        float lo = 0.f, hi = 0.f;
        if (ng < NE) {
            const float2 p = *(const float2*)(E + (size_t)ng * Dd + k2);
            lo = p.x; hi = p.y;
        }
        *(unsigned int*)(ldsB + n * Dd + k2) = cvt_pk_bf16(lo, hi);
    }
    __syncthreads();

    FragC acc[8];
#pragma unroll
    for (int s = 0; s < 8; ++s)
#pragma unroll
        for (int i = 0; i < 8; ++i) acc[s].f[i] = 0.f;

    const int ml = (wave << 4) + (lane & 15);
    const unsigned short* xrow = x1 + (size_t)(mbase + ml) * Dd; // L2-resident
    const int ahi = (lane & 16) ? 8 : 0;
    const int bhi = (lane & 16) ? 8 : 0;

#pragma unroll
    for (int kk = 0; kk < 8; ++kk) {
        const int k0 = kk << 5;
        FragA a;
        const uint4* ap = (const uint4*)(xrow + k0 + ahi);
        a.q[0] = ap[0];
        a.q[1] = ap[2];
        // two groups of 4: batch DS loads, then WMMAs (amortize dscnt waits)
#pragma unroll
        for (int g = 0; g < 2; ++g) {
            FragA bq[4];
#pragma unroll
            for (int s = 0; s < 4; ++s) {
                const uint4* bp = (const uint4*)(
                    ldsB + ((g * 4 + s) * 16 + (lane & 15)) * Dd + k0 + bhi);
                bq[s].q[0] = bp[0]; bq[s].q[1] = bp[2];
            }
#pragma unroll
            for (int s = 0; s < 4; ++s)
                acc[g * 4 + s].v = __builtin_amdgcn_wmma_f32_16x16x32_bf16(
                    false, a.v, false, bq[s].v, (short)0, acc[g * 4 + s].v, false, false);
        }
    }

    // Fused sigmoid + coalesced store (dominant HBM traffic: 410 MB)
#pragma unroll
    for (int s = 0; s < 8; ++s) {
        int n = nbase + s * 16 + (lane & 15);
        if (n < NE) {
#pragma unroll
            for (int v = 0; v < 8; ++v) {
                int m = mbase + (wave << 4) + v + ((lane & 16) ? 8 : 0);
                float x = acc[s].f[v];
                out[(size_t)m * NE + n] = 1.f / (1.f + __expf(-x));
            }
        }
    }
}

// ---------------- launch ----------------
extern "C" void kernel_launch(void* const* d_in, const int* in_sizes, int n_in,
                              void* d_out, int out_size, void* d_ws, size_t ws_size,
                              hipStream_t stream) {
    const int*   e1  = (const int*)d_in[0];
    const int*   rel = (const int*)d_in[1];
    const float* E   = (const float*)d_in[2];
    const float* R   = (const float*)d_in[3];
    const float* W   = (const float*)d_in[4];
    const float* g0  = (const float*)d_in[5];
    const float* b0  = (const float*)d_in[6];
    const float* g1  = (const float*)d_in[7];
    const float* b1  = (const float*)d_in[8];
    float* out = (float*)d_out;
    const int NE = in_sizes[2] / Dd;          // 100000

    // workspace carve-up (fp32 units; all 16B aligned)
    float* ws     = (float*)d_ws;
    float* scale0 = ws;                        // 256
    float* shift0 = ws + 256;                  // 256
    float* scale1 = ws + 512;                  // 256
    float* shift1 = ws + 768;                  // 256
    float* out1   = ws + 1024;                 // 1024*256
    float* rbuf   = out1 + (size_t)Bsz * Dd;   // 1024*256
    float* out1p  = rbuf + (size_t)Bsz * Dd;   // 8*1024*256 split-K partials
    unsigned short* x0 = (unsigned short*)(out1p + (size_t)8 * Bsz * Dd); // bf16
    unsigned short* x1 = x0 + (size_t)Bsz * Dd;                            // bf16

    prep0_stats<<<1, 256, 0, stream>>>(e1, E, g0, b0, scale0, shift0);
    prep0_apply<<<Bsz, 256, 0, stream>>>(e1, rel, E, R, scale0, shift0, x0, rbuf);

    // M-tiles 8 x N-tiles 4 x splitK 8 = 256 workgroups
    gemm1_kernel<<<dim3(4, 8, 8), 256, (128 * Dd) * 2 + 2 * (64 * 32) * 2, stream>>>(
        W, x0, rbuf, out1p);

    bn1_stats<<<1, 256, 0, stream>>>(out1p, g1, b1, out1, scale1, shift1);
    bn1_apply<<<Bsz, 256, 0, stream>>>(out1, scale1, shift1, x1);

    gemm2_kernel<<<dim3((NE + 127) / 128, 8), 256, 128 * Dd * 2, stream>>>(
        E, x1, out, NE);
}
